// SpatioTemporalAttention_75977971466367
// MI455X (gfx1250) — matrix operand order repaired and checked by reading, not compiled, not run on hardware
//
#include <hip/hip_runtime.h>
#include <hip/hip_bf16.h>

// ---------------------------------------------------------------------------
// SpatioTemporalAttention for MI455X (gfx1250, wave32, WMMA)
//   B=32, C=256, H=W=32, N=1024
//   f32 -> f16 operands, v_wmma_f32_16x16x32_f16 with f32 accumulation.
// ---------------------------------------------------------------------------

typedef __attribute__((ext_vector_type(16))) _Float16 v16h;
typedef __attribute__((ext_vector_type(8)))  _Float16 v8h;
typedef __attribute__((ext_vector_type(8)))  float    v8f;
typedef __attribute__((ext_vector_type(4)))  float    v4f;

#define BATCH 32
#define CCH   256
#define NPOS  1024
#define SCALE 0.0625f   // 1/sqrt(256)

// ---- WMMA wrapper ---------------------------------------------------------
__device__ __forceinline__ v8f wmma_f16(v16h a, v16h b, v8f c) {
  return __builtin_amdgcn_wmma_f32_16x16x32_f16(
      /*neg_a=*/false, a, /*neg_b=*/false, b,
      /*c_mod=*/(short)0, c, /*reuse_a=*/false, /*reuse_b=*/false);
}

// ---- A fragment: 16(M) x 32(K) f16, row-major source, ld in halves --------
// lane: M = lane&15 ; K base = (lane>>4)*8 ; halves 0..7 -> K kb..kb+7,
// halves 8..15 -> K kb+16..kb+23  (ISA 05_wmma 7.12.2)
__device__ __forceinline__ v16h load_a_frag(const _Float16* base, int ld) {
  int lane = threadIdx.x & 31;
  const _Float16* p = base + (lane & 15) * ld + ((lane >> 4) << 3);
  v8h lo = *(const v8h*)(p);
  v8h hi = *(const v8h*)(p + 16);
  v16h r;
#pragma unroll
  for (int i = 0; i < 8; ++i) { r[i] = lo[i]; r[i + 8] = hi[i]; }
  return r;
}

// ---- B fragment: 32(K) x 16(Ncol) f16 from a *transposed* source ----------
// Source is Bt[col][k] row-major with leading dim ld (halves).
// lane: col = lane&15 ; K base = (lane>>4)*8 ; halves 0..7 -> K kb..kb+7,
// halves 8..15 -> K kb+16..kb+23 (same as A fragment).
__device__ __forceinline__ v16h load_bT_frag(const _Float16* baseT, int ld) {
  int lane = threadIdx.x & 31;
  const _Float16* p = baseT + (lane & 15) * ld + ((lane >> 4) << 3);
  v8h lo = *(const v8h*)(p);
  v8h hi = *(const v8h*)(p + 16);
  v16h r;
#pragma unroll
  for (int i = 0; i < 8; ++i) { r[i] = lo[i]; r[i + 8] = hi[i]; }
  return r;
}

// ---------------------------------------------------------------------------
// Kernel 0: convert f32 weights -> f16 (Wq|Wk|Wv concatenated in ws)
// ---------------------------------------------------------------------------
__global__ void cvt_weights(const float* __restrict__ wq,
                            const float* __restrict__ wk,
                            const float* __restrict__ wv,
                            _Float16* __restrict__ w16) {
  int i = blockIdx.x * 256 + threadIdx.x;
  if (i < CCH * CCH) {
    w16[i]                 = (_Float16)wq[i];
    w16[CCH * CCH + i]     = (_Float16)wk[i];
    w16[2 * CCH * CCH + i] = (_Float16)wv[i];
  }
}

// ---------------------------------------------------------------------------
// Kernel 1: QKV projection.  grid = (N/64, B, 3), block = 256 (8 waves).
//   sel 0: Q -> Qt[b][n][c] (f16, transposed)
//   sel 1: K -> Kt[b][n][c]
//   sel 2: V -> Vh[b][c][n]   (coalesced via LDS bounce)
// ---------------------------------------------------------------------------
#define XS_LD 264   // 256 + 8 pad (halves)
#define VT_LD 64    // V bounce tile leading dim (halves)

__global__ void qkv_kernel(const float* __restrict__ x,
                           const _Float16* __restrict__ w16,
                           const float* __restrict__ bq,
                           const float* __restrict__ bk,
                           const float* __restrict__ bv,
                           _Float16* __restrict__ Qt,
                           _Float16* __restrict__ Kt,
                           _Float16* __restrict__ Vh) {
  __shared__ _Float16 Xs[64 * XS_LD];   // x-tile transposed: [n][c] f16

  const int n0  = blockIdx.x * 64;
  const int b   = blockIdx.y;
  const int sel = blockIdx.z;
  const int tid = threadIdx.x;

  const _Float16* W = w16 + sel * CCH * CCH;
  const float* bias = (sel == 0) ? bq : ((sel == 1) ? bk : bv);

  // Stage x[b][c][n0:n0+64) into LDS as f16 [n][c]  (thread t == channel t)
  {
    const float* xp = x + ((size_t)b * CCH + tid) * NPOS + n0;
#pragma unroll
    for (int j = 0; j < 64; j += 4) {
      v4f v = *(const v4f*)(xp + j);
#pragma unroll
      for (int i = 0; i < 4; ++i)
        Xs[(j + i) * XS_LD + tid] = (_Float16)v[i];
    }
  }
  __syncthreads();

  const int wave = tid >> 5, lane = tid & 31;
  const int mbase = wave * 32;          // 2 M-tiles per wave

  v8f acc[2][4];
#pragma unroll
  for (int i = 0; i < 2; ++i)
#pragma unroll
    for (int j = 0; j < 4; ++j) acc[i][j] = (v8f)(0.0f);

#pragma unroll
  for (int kk = 0; kk < 8; ++kk) {      // K = 256 in steps of 32
    v16h a0 = load_a_frag(W + (size_t)mbase * CCH + kk * 32, CCH);
    v16h a1 = load_a_frag(W + (size_t)(mbase + 16) * CCH + kk * 32, CCH);
    v16h bf[4];
#pragma unroll
    for (int nt = 0; nt < 4; ++nt)
      bf[nt] = load_bT_frag(Xs + (nt * 16) * XS_LD + kk * 32, XS_LD);
#pragma unroll
    for (int nt = 0; nt < 4; ++nt) {
      acc[0][nt] = wmma_f16(a0, bf[nt], acc[0][nt]);
      acc[1][nt] = wmma_f16(a1, bf[nt], acc[1][nt]);
    }
  }

  // Epilogue: +bias, cvt f16, store.
  const int lg = lane >> 4, li = lane & 15;
  if (sel < 2) {
    _Float16* outp = (sel == 0 ? Qt : Kt);
#pragma unroll
    for (int mi = 0; mi < 2; ++mi) {
      const int o0 = mbase + mi * 16 + lg * 8;   // 8 contiguous out-channels
      v4f bA = *(const v4f*)(bias + o0);
      v4f bB = *(const v4f*)(bias + o0 + 4);
#pragma unroll
      for (int nt = 0; nt < 4; ++nt) {
        const int n = n0 + nt * 16 + li;
        v8f a = acc[mi][nt];
        v8h pk;
#pragma unroll
        for (int r = 0; r < 4; ++r) pk[r] = (_Float16)(a[r] + bA[r]);
#pragma unroll
        for (int r = 0; r < 4; ++r) pk[r + 4] = (_Float16)(a[r + 4] + bB[r]);
        *(v8h*)(outp + ((size_t)b * NPOS + n) * CCH + o0) = pk;
      }
    }
  } else {
    // V: bounce through LDS (reusing Xs) for full-cacheline global stores.
    __syncthreads();                       // all waves done reading Xs
    _Float16* Vt = Xs;                     // [256 c][64 n] f16 tile
#pragma unroll
    for (int mi = 0; mi < 2; ++mi) {
      const int o0 = mbase + mi * 16 + lg * 8;
      v4f bA = *(const v4f*)(bias + o0);
      v4f bB = *(const v4f*)(bias + o0 + 4);
#pragma unroll
      for (int nt = 0; nt < 4; ++nt) {
        const int nc = nt * 16 + li;
        v8f a = acc[mi][nt];
#pragma unroll
        for (int r = 0; r < 8; ++r) {
          float bb = (r < 4) ? bA[r] : bB[r - 4];
          Vt[(o0 + r) * VT_LD + nc] = (_Float16)(a[r] + bb);
        }
      }
    }
    __syncthreads();
    // thread t == channel t: 64 contiguous halves (128 B) -> HBM
    const _Float16* src = Vt + tid * VT_LD;
    _Float16* dst = Vh + ((size_t)b * CCH + tid) * NPOS + n0;
#pragma unroll
    for (int k = 0; k < 64; k += 8)
      *(v8h*)(dst + k) = *(const v8h*)(src + k);
  }
}

// ---------------------------------------------------------------------------
// Kernel 2: fused attention.  grid = (N/32, B), block = 256 (8 waves).
// One block owns a 32-row tile: S [32][1024] f16 in LDS (in-place softmax),
// then P·V^T with residual.  Halves K/V L2 traffic vs 16-row tiles.
// ---------------------------------------------------------------------------
#define QS_LD  264          // halves
#define S16_LD 1032         // halves (1024 + 8 pad)
#define QS_BYTES  (32 * QS_LD * 2)     // 16896
#define S_BYTES   (32 * S16_LD * 2)    // 66048
#define RED_BYTES (32 * 8 * 4)         // 1024
#define SMEM2 (QS_BYTES + S_BYTES + RED_BYTES + 128)

__global__ void attn_kernel(const float* __restrict__ x,
                            const _Float16* __restrict__ Qt,
                            const _Float16* __restrict__ Kt,
                            const _Float16* __restrict__ Vh,
                            float* __restrict__ out) {
  extern __shared__ char smem[];
  _Float16* Qs  = (_Float16*)smem;                         // [32][QS_LD]
  _Float16* Sh  = (_Float16*)(smem + QS_BYTES);            // [32][S16_LD]
  float*    red = (float*)(smem + QS_BYTES + S_BYTES);     // [32][8]
  float*    rowinv = (float*)(smem + QS_BYTES + S_BYTES + RED_BYTES); // [32]

  const int ntile = blockIdx.x;      // 32-row tile of N
  const int b     = blockIdx.y;
  const int tid   = threadIdx.x;
  const int wave  = tid >> 5, lane = tid & 31;
  const int lg = lane >> 4, li = lane & 15;

  // ---- 1. stage Q-tile [32][256] f16 into LDS ----
  {
    const int r  = tid >> 3;           // row 0..31
    const int cp = (tid & 7) * 32;     // 32 halves per thread
    const _Float16* qp = Qt + ((size_t)b * NPOS + ntile * 32 + r) * CCH + cp;
#pragma unroll
    for (int k = 0; k < 32; k += 8)
      *(v8h*)(Qs + r * QS_LD + cp + k) = *(const v8h*)(qp + k);
  }
  __syncthreads();

  // ---- 2. S = Q^T K : 64 m-tiles, 8 per wave, 2 row-subtiles each ----
  for (int j = 0; j < 8; ++j) {
    const int mt = j * 8 + wave;
    v8f acc0 = (v8f)(0.0f), acc1 = (v8f)(0.0f);
    const _Float16* kbase = Kt + ((size_t)b * NPOS + mt * 16) * CCH;
#pragma unroll
    for (int kk = 0; kk < 8; ++kk) {
      v16h bb = load_bT_frag(kbase + kk * 32, CCH);
      v16h a0 = load_a_frag(Qs + kk * 32, QS_LD);
      v16h a1 = load_a_frag(Qs + 16 * QS_LD + kk * 32, QS_LD);
      acc0 = wmma_f16(a0, bb, acc0);
      acc1 = wmma_f16(a1, bb, acc1);
    }
    const int rr  = lg * 8;
    const int col = mt * 16 + li;
#pragma unroll
    for (int r = 0; r < 8; ++r) {
      Sh[(rr + r) * S16_LD + col]        = (_Float16)acc0[r];
      Sh[(16 + rr + r) * S16_LD + col]   = (_Float16)acc1[r];
    }
  }
  __syncthreads();

  // ---- 3. row softmax over 1024 cols (8 threads per row), in place ----
  {
    const int row = tid >> 3, sub = tid & 7;
    float mx = -1e30f;
    for (int j = sub; j < NPOS; j += 8)
      mx = fmaxf(mx, (float)Sh[row * S16_LD + j]);
    red[row * 8 + sub] = mx;
    __syncthreads();
    float rmax = -1e30f;
#pragma unroll
    for (int i = 0; i < 8; ++i) rmax = fmaxf(rmax, red[row * 8 + i]);
    __syncthreads();                       // red about to be reused
    float sum = 0.0f;
    for (int j = sub; j < NPOS; j += 8) {
      float e = __expf(SCALE * ((float)Sh[row * S16_LD + j] - rmax));
      sum += e;
      Sh[row * S16_LD + j] = (_Float16)e;  // in place: P overwrites S
    }
    red[row * 8 + sub] = sum;
    __syncthreads();
    float rs = 0.0f;
#pragma unroll
    for (int i = 0; i < 8; ++i) rs += red[row * 8 + i];
    if (sub == 0) rowinv[row] = 1.0f / rs;
  }
  __syncthreads();

  // ---- 4. O = P · V^T : each wave owns 32 channels x 32 rows ----
  v8f oacc[2][2];        // [nt2][ct]
#pragma unroll
  for (int i = 0; i < 2; ++i)
#pragma unroll
    for (int j2 = 0; j2 < 2; ++j2) oacc[i][j2] = (v8f)(0.0f);

  const int c0 = wave * 32;
  const _Float16* vbase = Vh + (size_t)b * CCH * NPOS;
  for (int kk = 0; kk < 32; ++kk) {
    v16h b0 = load_bT_frag(vbase + (size_t)c0 * NPOS + kk * 32, NPOS);
    v16h b1 = load_bT_frag(vbase + (size_t)(c0 + 16) * NPOS + kk * 32, NPOS);
    v16h a0 = load_a_frag(Sh + kk * 32, S16_LD);
    v16h a1 = load_a_frag(Sh + 16 * S16_LD + kk * 32, S16_LD);
    oacc[0][0] = wmma_f16(a0, b0, oacc[0][0]);
    oacc[0][1] = wmma_f16(a0, b1, oacc[0][1]);
    oacc[1][0] = wmma_f16(a1, b0, oacc[1][0]);
    oacc[1][1] = wmma_f16(a1, b1, oacc[1][1]);
  }

  // ---- 5. epilogue: *1/rowsum, +residual, vector f32 stores ----
  const int nrow0 = lg * 8;
#pragma unroll
  for (int nt2 = 0; nt2 < 2; ++nt2) {
#pragma unroll
    for (int ct = 0; ct < 2; ++ct) {
      const int c = c0 + ct * 16 + li;
      const size_t base = ((size_t)b * CCH + c) * NPOS +
                          (size_t)ntile * 32 + nt2 * 16 + nrow0;
      v8f a = oacc[nt2][ct];
      v4f x0 = *(const v4f*)(x + base);
      v4f x1 = *(const v4f*)(x + base + 4);
      v4f o0, o1;
#pragma unroll
      for (int r = 0; r < 4; ++r) {
        o0[r] = a[r]     * rowinv[nt2 * 16 + nrow0 + r]     + x0[r];
        o1[r] = a[r + 4] * rowinv[nt2 * 16 + nrow0 + 4 + r] + x1[r];
      }
      *(v4f*)(out + base)     = o0;
      *(v4f*)(out + base + 4) = o1;
    }
  }
}

// ---------------------------------------------------------------------------
// Launch
// ---------------------------------------------------------------------------
extern "C" void kernel_launch(void* const* d_in, const int* in_sizes, int n_in,
                              void* d_out, int out_size, void* d_ws, size_t ws_size,
                              hipStream_t stream) {
  const float* x  = (const float*)d_in[0];
  const float* Wq = (const float*)d_in[1];
  const float* bq = (const float*)d_in[2];
  const float* Wk = (const float*)d_in[3];
  const float* bk = (const float*)d_in[4];
  const float* Wv = (const float*)d_in[5];
  const float* bv = (const float*)d_in[6];
  float* out = (float*)d_out;

  // Workspace layout (f16):
  //   [0)          W16: 3 * 256*256          = 384 KB
  //   [WOFF)       Qt : B*N*C (transposed)   = 16 MB
  //   [WOFF+QSZ)   Kt : B*N*C (transposed)   = 16 MB
  //   [WOFF+2*QSZ) Vh : B*C*N                = 16 MB
  const size_t WOFF = (size_t)3 * CCH * CCH * sizeof(_Float16);
  const size_t QSZ  = (size_t)BATCH * NPOS * CCH * sizeof(_Float16);
  _Float16* W16 = (_Float16*)d_ws;
  _Float16* Qt  = (_Float16*)((char*)d_ws + WOFF);
  _Float16* Kt  = (_Float16*)((char*)d_ws + WOFF + QSZ);
  _Float16* Vh  = (_Float16*)((char*)d_ws + WOFF + 2 * QSZ);

  cvt_weights<<<(CCH * CCH + 255) / 256, 256, 0, stream>>>(Wq, Wk, Wv, W16);

  qkv_kernel<<<dim3(NPOS / 64, BATCH, 3), 256, 0, stream>>>(
      x, W16, bq, bk, bv, Qt, Kt, Vh);

  attn_kernel<<<dim3(NPOS / 32, BATCH), 256, SMEM2, stream>>>(
      x, Qt, Kt, Vh, out);
}
